// RNN_10118942949526
// MI455X (gfx1250) — matrix operand order repaired and checked by reading, not compiled, not run on hardware
//
#include <hip/hip_runtime.h>
#include <hip/hip_bf16.h>

// ---- CDNA5 WMMA types --------------------------------------------------
typedef __attribute__((ext_vector_type(16))) __bf16 v16bf;
typedef __attribute__((ext_vector_type(8)))  float  v8f;
typedef __attribute__((ext_vector_type(4)))  int    v4i;

#define SEQ 512
#define BB  64
#define EE  512
#define HH  512
#define MTOT (SEQ*BB)     // 32768 rows of the big GEMM
#define KCH  (EE/32)      // 16 K-chunks of 32

// ---- CDNA5 async-load-to-LDS path (guarded; falls back clean) ----------
#if defined(__has_builtin)
#if __has_builtin(__builtin_amdgcn_global_load_async_to_lds_b128) && \
    __has_builtin(__builtin_amdgcn_s_wait_asynccnt)
#define USE_ASYNC 1
#endif
#endif

#if defined(USE_ASYNC)
typedef __attribute__((address_space(1))) v4i* gv4i_p;   // global v4i*
typedef __attribute__((address_space(3))) v4i* lv4i_p;   // LDS v4i*
#define WAIT_ASYNC(n) __builtin_amdgcn_s_wait_asynccnt(n)
__device__ __forceinline__ void cp16(__bf16* l, const __bf16* g) {
  __builtin_amdgcn_global_load_async_to_lds_b128((gv4i_p)g, (lv4i_p)l, 0, 0);
}
#else
#define WAIT_ASYNC(n)
__device__ __forceinline__ void cp16(__bf16* l, const __bf16* g) {
  *(uint4*)l = *(const uint4*)g;
}
#endif

// ---- A-fragment packed layout (16-bit A-matrix 16x32, wave32) ----------
// frag element j of lane:  p=j>>1,  K = (p&3)*2 + (p>>2)*16 + (lane>>4)*8 + (j&1)
__device__ __forceinline__ int packA_idx(int m, int k) {
  int mt = m >> 4, lm = m & 15;
  int kc = k >> 5, kin = k & 31;
  int lh = (kin >> 3) & 1;
  int kk = kin - (lh << 3);              // in {0..7} u {16..23}
  int p  = ((kk >> 1) & 3) | ((kk >> 4) << 2);
  int j  = (p << 1) | (kk & 1);
  int lane = lm | (lh << 4);
  return ((mt * KCH + kc) * 32 + lane) * 16 + j;
}

__device__ __forceinline__ v8f wmma_bf16(v16bf a, v16bf b, v8f c) {
  return __builtin_amdgcn_wmma_f32_16x16x32_bf16(false, a, false, b, (short)0, c,
                                                 false, false);
}

// ------------------------------------------------------------------------
// Kernel 0a: pack Wx / Wh into bf16 B-fragment layout
//   Bpack[nt][kc][lane][j]:  n = nt*16+(lane&15), k = kc*32 + (j&7) + 8*(lane>>4) + 16*(j>>3)
// Also: h0 -> A-fragment-packed bf16 ping buffer, zero barrier counter.
// ------------------------------------------------------------------------
__global__ __launch_bounds__(256)
void pack_kernel(const float* __restrict__ W_enc,
                 const float* __restrict__ h0,
                 __bf16* __restrict__ WxP,
                 __bf16* __restrict__ WhP,
                 __bf16* __restrict__ hbuf0,
                 unsigned int* __restrict__ ctr) {
  int idx = blockIdx.x * blockDim.x + threadIdx.x;   // over H*E = 512*512
  if (idx < HH * EE) {
    int n   = idx >> 9;
    int k   = idx & 511;
    int nt  = n >> 4, lnN = n & 15;
    int kc  = k >> 5, kin = k & 31;
    int lane = lnN + (((kin >> 3) & 1) << 4);
    int j    = (kin & 7) + ((kin >> 4) << 3);
    size_t p = (((size_t)nt * KCH + kc) * 32 + lane) * 16 + j;
    const float* wrow = W_enc + (size_t)n * (EE + HH);
    WxP[p] = (__bf16)wrow[k];
    WhP[p] = (__bf16)wrow[EE + k];
  }
  if (idx < BB * HH) {
    int m = idx >> 9, n = idx & 511;
    hbuf0[packA_idx(m, n)] = (__bf16)h0[idx];
  }
  if (idx == 0) *ctr = 0u;
}

// ------------------------------------------------------------------------
// Kernel 0b: x (f32) -> A-fragment-packed bf16 (gather-style: coalesced
// 2B writes, cache-friendly 32B-line reads).  32 MB result is L2-resident.
// ------------------------------------------------------------------------
__global__ __launch_bounds__(256)
void xpack_kernel(const float* __restrict__ x, __bf16* __restrict__ xP) {
  size_t idx = (size_t)blockIdx.x * blockDim.x + threadIdx.x;  // MTOT*EE
  int j    = (int)(idx & 15);
  int lane = (int)((idx >> 4) & 31);
  size_t rest = idx >> 9;
  int kc = (int)(rest & (KCH - 1));
  size_t mt = rest >> 4;
  int p = j >> 1;
  int k = kc * 32 + ((p & 3) << 1) + ((p >> 2) << 4) + ((lane >> 4) << 3) + (j & 1);
  size_t m = mt * 16 + (lane & 15);
  xP[idx] = (__bf16)x[m * EE + k];
}

// ------------------------------------------------------------------------
// Kernel 1: xw[s,b,h] = sum_e x[s,b,e]*Wx[h,e] + b_enc[h]
// Block = 256 thr (8 waves) -> 128x64 tile.  A tiles (fragment-packed
// bf16) staged into LDS with async-to-LDS, double buffered 2 deep; all
// four B fragments issued before the WMMAs so loads overlap matrix math.
// ------------------------------------------------------------------------
__global__ __launch_bounds__(256)
void xw_gemm_kernel(const __bf16* __restrict__ xP,
                    const __bf16* __restrict__ WxP,
                    const float* __restrict__ b_enc,
                    float* __restrict__ xw) {
  __shared__ __align__(32) __bf16 As[2][8 * 512];   // 2 x 8 KB
  const int tid = threadIdx.x;
  const int wave = tid >> 5, lane = tid & 31;
  const int laneLo = lane & 15, laneHi = lane >> 4;
  const size_t mt0 = (size_t)blockIdx.x * 8;        // 8 row-tiles of 16
  const int n0 = blockIdx.y * 64;

  const int w8 = tid >> 5, l32 = tid & 31;          // staging roles
  auto stage = [&](int s, int kc) {
    const __bf16* g = xP + ((mt0 + w8) * KCH + kc) * 512 + l32 * 16;
    __bf16* l = &As[s][w8 * 512 + l32 * 16];
    cp16(l, g);
    cp16(l + 8, g + 8);
  };

  const v8f vzero = {0.f, 0.f, 0.f, 0.f, 0.f, 0.f, 0.f, 0.f};
  v8f acc[4];
#pragma unroll
  for (int c = 0; c < 4; ++c) acc[c] = vzero;

  const __bf16* bBase = WxP + (((size_t)(n0 >> 4) * KCH) * 32 + lane) * 16;

  auto compute = [&](int kc, int s) {
    const v16bf a = *(const v16bf*)(&As[s][wave * 512 + lane * 16]);
    const __bf16* bp = bBase + (size_t)kc * 512;    // 32 lanes * 16 halfs
    // issue ALL fragment loads first, then consume -> staggered waits
    const v16bf b0 = *(const v16bf*)(bp);
    const v16bf b1 = *(const v16bf*)(bp + (size_t)1 * KCH * 512);
    const v16bf b2 = *(const v16bf*)(bp + (size_t)2 * KCH * 512);
    const v16bf b3 = *(const v16bf*)(bp + (size_t)3 * KCH * 512);
    acc[0] = wmma_bf16(a, b0, acc[0]);
    acc[1] = wmma_bf16(a, b1, acc[1]);
    acc[2] = wmma_bf16(a, b2, acc[2]);
    acc[3] = wmma_bf16(a, b3, acc[3]);
  };

  stage(0, 0);
  stage(1, 1);
  for (int kc = 0; kc < KCH - 1; ++kc) {
    WAIT_ASYNC(2);                 // oldest in-flight stage complete
    __syncthreads();
    compute(kc, kc & 1);
    __syncthreads();
    if (kc + 2 < KCH) stage(kc & 1, kc + 2);
  }
  WAIT_ASYNC(0);
  __syncthreads();
  compute(KCH - 1, (KCH - 1) & 1);

  // epilogue: C/D layout — VGPR r = row M=r (lanes 0-15) / r+8 (lanes 16-31)
#pragma unroll
  for (int c = 0; c < 4; ++c) {
    const int col = n0 + c * 16 + laneLo;
    const float bias = b_enc[col];
#pragma unroll
    for (int r = 0; r < 8; ++r) {
      const size_t m = mt0 * 16 + wave * 16 + r + laneHi * 8;
      xw[m * HH + col] = acc[c][r] + bias;
    }
  }
}

// ------------------------------------------------------------------------
// Kernel 2: sequential scan, 16 persistent blocks (cluster-sized).
// h kept fragment-packed bf16 (double buffered, L2-resident); the serial
// inner loop is software-pipelined {load kc+1 | wmma kc} with no barriers.
// Per-step sync: fence + s_cluster_barrier + atomic counter.
// ------------------------------------------------------------------------
__global__ __launch_bounds__(256)
void rnn_scan_kernel(const float* __restrict__ xw,
                     const __bf16* __restrict__ WhP,
                     __bf16* __restrict__ hbufP,     // 2 * B*H bf16 (packed)
                     float* __restrict__ out,
                     unsigned int* __restrict__ ctr) {
  const int tid = threadIdx.x;
  const int wave = tid >> 5, lane = tid & 31;
  const int laneLo = lane & 15, laneHi = lane >> 4;
  const int n0 = blockIdx.x * 32;            // block's 32 h-columns
  const int mt = wave & 3;                   // row-tile 0..3 (M=64)
  const int m0 = mt * 16;
  const int nw = n0 + (wave >> 2) * 16;
  const int nt = nw >> 4;
  volatile unsigned int* vctr = ctr;

  const __bf16* bBase = WhP + (((size_t)nt * KCH) * 32 + lane) * 16;

  for (int t = 0; t < SEQ; ++t) {
    const __bf16* hprev = hbufP + (size_t)(t & 1) * (BB * HH);
    __bf16*       hnext = hbufP + (size_t)((t + 1) & 1) * (BB * HH);
    const __bf16* aBase = hprev + (size_t)mt * KCH * 512 + lane * 16;

    // accumulator initialized with precomputed xw_t (f32)
    v8f acc;
    const float* xwt = xw + (size_t)t * BB * HH;
#pragma unroll
    for (int r = 0; r < 8; ++r)
      acc[r] = xwt[(size_t)(m0 + r + laneHi * 8) * HH + nw + laneLo];

    if (t + 1 < SEQ)   // warm L2/GL1 for next step's xw slice
      __builtin_prefetch(xw + (size_t)(t + 1) * BB * HH + (size_t)(m0 + laneHi * 8) * HH + nw + laneLo, 0, 1);

    // software-pipelined K loop: preload next fragments before each WMMA
    v16bf a = *(const v16bf*)(aBase);
    v16bf b = *(const v16bf*)(bBase);
#pragma unroll
    for (int kc = 0; kc < KCH - 1; ++kc) {
      const v16bf an = *(const v16bf*)(aBase + (size_t)(kc + 1) * 512);
      const v16bf bn = *(const v16bf*)(bBase + (size_t)(kc + 1) * 512);
      acc = wmma_bf16(a, b, acc);
      a = an; b = bn;
    }
    acc = wmma_bf16(a, b, acc);

    // tanh + write hidden state (f32 out, packed bf16 next-state)
#pragma unroll
    for (int r = 0; r < 8; ++r) {
      const int m = m0 + r + laneHi * 8;
      const int n = nw + laneLo;
      const float hv = tanhf(acc[r]);
      out[((size_t)t * BB + m) * HH + n] = hv;
      hnext[packA_idx(m, n)] = (__bf16)hv;
    }

    // ---- inter-workgroup step barrier ----
    __threadfence();                       // release h writes (device scope)
    __syncthreads();
    __builtin_amdgcn_s_cluster_barrier();  // HW cluster barrier (NOP if uncl.)
    if (tid == 0) {
      atomicAdd(ctr, 1u);
      const unsigned int target = 16u * (unsigned int)(t + 1);
      while (*vctr < target) { __builtin_amdgcn_s_sleep(1); }
    }
    __syncthreads();
    __threadfence();                       // acquire other blocks' h writes
  }
}

// ------------------------------------------------------------------------
extern "C" void kernel_launch(void* const* d_in, const int* in_sizes, int n_in,
                              void* d_out, int out_size, void* d_ws, size_t ws_size,
                              hipStream_t stream) {
  const float* x     = (const float*)d_in[0];  // (SEQ,B,E)
  const float* h0    = (const float*)d_in[1];  // (B,H)
  const float* W_enc = (const float*)d_in[2];  // (H,E+H)
  const float* b_enc = (const float*)d_in[3];  // (H,)
  float* out = (float*)d_out;                  // (SEQ,B,H)

  char* ws = (char*)d_ws;
  float* xw = (float*)ws;                                      // 64 MB
  size_t off = (size_t)MTOT * HH * sizeof(float);
  __bf16* xP  = (__bf16*)(ws + off); off += (size_t)MTOT * EE * sizeof(__bf16);  // 32 MB
  __bf16* WxP = (__bf16*)(ws + off); off += (size_t)HH * EE * sizeof(__bf16);
  __bf16* WhP = (__bf16*)(ws + off); off += (size_t)HH * HH * sizeof(__bf16);
  __bf16* hbuf = (__bf16*)(ws + off); off += 2u * BB * HH * sizeof(__bf16);
  unsigned int* ctr = (unsigned int*)(ws + off);

  pack_kernel<<<(HH * EE + 255) / 256, 256, 0, stream>>>(W_enc, h0, WxP, WhP, hbuf, ctr);
  xpack_kernel<<<(int)(((size_t)MTOT * EE + 255) / 256), 256, 0, stream>>>(x, xP);

  dim3 g1(MTOT / 128, HH / 64);   // 256 x 8 blocks
  xw_gemm_kernel<<<g1, 256, 0, stream>>>(xP, WxP, b_enc, xw);

  rnn_scan_kernel<<<16, 256, 0, stream>>>(xw, WhP, hbuf, out, ctr);
}
